// CustomLoss_35940286333129
// MI455X (gfx1250) — matrix operand order repaired and checked by reading, not compiled, not run on hardware
//
#include <hip/hip_runtime.h>
#include <hip/hip_bf16.h>

// ---------------------------------------------------------------------------
// CDNA5 (gfx1250) composite loss:  out[b] = mae + chamfer + emd[b]
// Chamfer's batched 1024x1024x128 GEMM runs on v_wmma_f32_16x16x32_bf16 with a
// PRE-SPLIT bf16-hi/lo decomposition (3 WMMAs per K-step, ~fp32 accuracy).
// The split is done once in a streaming prep pass so the WMMA loop contains
// only bf16 fragment loads + WMMAs (no per-tile cvt chains).
// ---------------------------------------------------------------------------

typedef __attribute__((ext_vector_type(16))) __bf16 v16bf;
typedef __attribute__((ext_vector_type(8)))  __bf16 v8bf;
typedef __attribute__((ext_vector_type(4)))  __bf16 v4bf;
typedef __attribute__((ext_vector_type(8)))  float  v8f;

#define B_    32
#define N_    1024
#define D_    128
#define NM_   (N_ * N_)        // 1048576
#define BN_   (B_ * N_)        // 32768
#define TOT_  (B_ * N_ * D_)   // 4194304
#define SORTN (N_ * D_)        // 131072 elements per batch (2^17)

// workspace layout (float units)
#define WS_SCAL     0                    // [0]=mae [1]=colmin_sum [2]=batchmin_sum [3..34]=emd[b]
#define WS_PNORM    64                   // 32768
#define WS_TNORM    (WS_PNORM + BN_)     // 32768
#define WS_COLMIN   (WS_TNORM + BN_)     // 32768 (uint bit patterns)
#define WS_BATCHMIN (WS_COLMIN + BN_)    // 1048576
#define WS_PSORT    (WS_BATCHMIN + NM_)  // 4194304
#define WS_GSORT    (WS_PSORT + TOT_)    // 4194304
#define WS_BF16     (WS_GSORT + TOT_)    // 4 bf16 arrays of TOT_ (phi,plo,thi,tlo)

__device__ __forceinline__ float wave_sum(float v) {
#pragma unroll
    for (int o = 16; o > 0; o >>= 1) v += __shfl_xor(v, o, 32);
    return v;
}

// ---------------------------------------------------------------------------
// init: zero scalar slots, +inf-fill colmin / batchmin
// ---------------------------------------------------------------------------
__global__ void k_init(float* ws) {
    const unsigned idx = blockIdx.x * blockDim.x + threadIdx.x;
    if (idx < 64) ws[WS_SCAL + idx] = 0.0f;
    unsigned int* colmin   = (unsigned int*)(ws + WS_COLMIN);
    unsigned int* batchmin = (unsigned int*)(ws + WS_BATCHMIN);
    if (idx < BN_) colmin[idx]   = 0x7f800000u;   // +inf
    if (idx < NM_) batchmin[idx] = 0x7f800000u;   // +inf
}

// ---------------------------------------------------------------------------
// row squared norms for pred and target ([B*N] each), fp32-exact
// ---------------------------------------------------------------------------
__global__ void k_norms(const float* __restrict__ pred,
                        const float* __restrict__ targ, float* ws) {
    const unsigned idx = blockIdx.x * blockDim.x + threadIdx.x;  // 0..2*BN_-1
    const bool is_t    = idx >= BN_;
    const unsigned r   = is_t ? (idx - BN_) : idx;
    const float4* row  = (const float4*)((is_t ? targ : pred) + (size_t)r * D_);
    float acc = 0.0f;
#pragma unroll 8
    for (int i = 0; i < D_ / 4; ++i) {
        float4 v = row[i];
        acc += v.x * v.x + v.y * v.y + v.z * v.z + v.w * v.w;
    }
    ws[(is_t ? WS_TNORM : WS_PNORM) + r] = acc;
}

// ---------------------------------------------------------------------------
// streaming prep: MAE partial sums + copy inputs into sort scratch +
// bf16 hi/lo split of both inputs (written once, reused by all 4096 tiles).
// One thread = 4 consecutive elements (float4 in, v4bf out).
// ---------------------------------------------------------------------------
__global__ void k_prep(const float* __restrict__ pred,
                       const float* __restrict__ targ, float* ws) {
    float*  psort = ws + WS_PSORT;
    float*  gsort = ws + WS_GSORT;
    __bf16* phi   = (__bf16*)(ws + WS_BF16);
    __bf16* plo   = phi + TOT_;
    __bf16* thi   = plo + TOT_;
    __bf16* tlo   = thi + TOT_;

    const unsigned i = (blockIdx.x * blockDim.x + threadIdx.x) * 4;  // exact
    float4 a = *(const float4*)(pred + i);
    float4 b = *(const float4*)(targ + i);
    *(float4*)(psort + i) = a;
    *(float4*)(gsort + i) = b;

    float av[4] = {a.x, a.y, a.z, a.w};
    float bv[4] = {b.x, b.y, b.z, b.w};
    v4bf ah, al, bh, bl;
    float acc = 0.0f;
#pragma unroll
    for (int e = 0; e < 4; ++e) {
        __bf16 h = (__bf16)av[e];
        ah[e] = h;
        al[e] = (__bf16)(av[e] - (float)h);
        __bf16 g = (__bf16)bv[e];
        bh[e] = g;
        bl[e] = (__bf16)(bv[e] - (float)g);
        acc += fabsf(av[e] - bv[e]);
    }
    *(v4bf*)(phi + i) = ah;
    *(v4bf*)(plo + i) = al;
    *(v4bf*)(thi + i) = bh;
    *(v4bf*)(tlo + i) = bl;

    acc = wave_sum(acc);
    if ((threadIdx.x & 31) == 0) atomicAdd(&ws[WS_SCAL + 0], acc);
}

// ---------------------------------------------------------------------------
// Chamfer distance tiles via WMMA bf16x3 on pre-split data.
// grid = (64, 16), block = 128 (4 waves); wave w handles tile (ti, bIdx.y*4+w)
// and loops all 32 batches (keeps the batch-axis min in registers).
// ---------------------------------------------------------------------------
__global__ void __launch_bounds__(128)
k_chamfer_wmma(float* ws) {
    const float*  pnorm    = ws + WS_PNORM;
    const float*  tnorm    = ws + WS_TNORM;
    unsigned int* colmin   = (unsigned int*)(ws + WS_COLMIN);
    float*        batchmin = ws + WS_BATCHMIN;
    const __bf16* phi      = (const __bf16*)(ws + WS_BF16);
    const __bf16* plo      = phi + TOT_;
    const __bf16* thi      = plo + TOT_;
    const __bf16* tlo      = thi + TOT_;

    const int lane = threadIdx.x & 31;
    const int wave = threadIdx.x >> 5;
    const int ti   = blockIdx.x;                 // row tile (pred)  0..63
    const int tj   = blockIdx.y * 4 + wave;      // col tile (targ) 0..63
    const int half = lane >> 4;
    const int l15  = lane & 15;

    const int rowA = ti * 16 + l15;              // pred row this lane loads (A)
    const int rowB = tj * 16 + l15;              // targ row this lane loads (B)
    const int col  = tj * 16 + l15;              // output column (C layout)

    float bm[8];
#pragma unroll
    for (int v = 0; v < 8; ++v) bm[v] = __builtin_inff();

    for (int b = 0; b < B_; ++b) {
        const __bf16* pa_hi = phi + ((size_t)b * N_ + rowA) * D_;
        const __bf16* pa_lo = plo + ((size_t)b * N_ + rowA) * D_;
        const __bf16* tb_hi = thi + ((size_t)b * N_ + rowB) * D_;
        const __bf16* tb_lo = tlo + ((size_t)b * N_ + rowB) * D_;
        v8f c = {};
#pragma unroll
        for (int ks = 0; ks < 4; ++ks) {
            // A (16x32 bf16): lanes 0-15 hold K=0..15, lanes 16-31 hold
            // K=16..31 -> two contiguous 8-element (16B) chunks.
            const int kA = ks * 32 + half * 16;
            v8bf a0 = *(const v8bf*)(pa_hi + kA);
            v8bf a1 = *(const v8bf*)(pa_hi + kA + 8);
            v8bf a2 = *(const v8bf*)(pa_lo + kA);
            v8bf a3 = *(const v8bf*)(pa_lo + kA + 8);
            v16bf ahi = __builtin_shufflevector(a0, a1, 0, 1, 2, 3, 4, 5, 6, 7,
                                                8, 9, 10, 11, 12, 13, 14, 15);
            v16bf alo = __builtin_shufflevector(a2, a3, 0, 1, 2, 3, 4, 5, 6, 7,
                                                8, 9, 10, 11, 12, 13, 14, 15);
            // B (32x16 bf16): lanes 0-15 K=0..15, lanes 16-31 K=16..31 ->
            // one contiguous 16-element (32B) chunk. (B[k][n] = targ[n][k])
            const int kB = ks * 32 + half * 16;
            v16bf bhi = *(const v16bf*)(tb_hi + kB);
            v16bf blo = *(const v16bf*)(tb_lo + kB);
            // bf16x3: hi*hi + hi*lo + lo*hi (~fp32 accuracy for the cross term)
            c = __builtin_amdgcn_wmma_f32_16x16x32_bf16(false, ahi, false, bhi,
                                                        (short)0, c, false, false);
            c = __builtin_amdgcn_wmma_f32_16x16x32_bf16(false, ahi, false, blo,
                                                        (short)0, c, false, false);
            c = __builtin_amdgcn_wmma_f32_16x16x32_bf16(false, alo, false, bhi,
                                                        (short)0, c, false, false);
        }
        // distances; C layout: c[v] -> (row = ti*16 + half*8 + v, col)
        const float tn = tnorm[b * N_ + col];
        float cmin = __builtin_inff();
#pragma unroll
        for (int v = 0; v < 8; ++v) {
            float pn = pnorm[b * N_ + ti * 16 + half * 8 + v];
            float sq = pn + tn - 2.0f * c[v];
            float dd = sqrtf(fmaxf(sq, 1e-12f));
            bm[v]    = fminf(bm[v], dd);
            cmin     = fminf(cmin, dd);
        }
        // per-column min across the two half-rows, then global per-(b,col) min
        cmin = fminf(cmin, __shfl_xor(cmin, 16, 32));
        if (half == 0)
            atomicMin(&colmin[b * N_ + col], __float_as_uint(cmin));
    }
    // min over the batch axis: tile (ti,tj) owned by exactly one wave -> store
#pragma unroll
    for (int v = 0; v < 8; ++v) {
        int row = ti * 16 + half * 8 + v;
        batchmin[row * N_ + col] = bm[v];
    }
}

// ---------------------------------------------------------------------------
// generic sum reduction: src[0..n) -> atomicAdd(*out)
// ---------------------------------------------------------------------------
__global__ void k_sum(const float* __restrict__ src, int n, float* out) {
    float acc = 0.0f;
    for (unsigned i = blockIdx.x * blockDim.x + threadIdx.x; i < (unsigned)n;
         i += gridDim.x * blockDim.x)
        acc += src[i];
    acc = wave_sum(acc);
    if ((threadIdx.x & 31) == 0) atomicAdd(out, acc);
}

// ---------------------------------------------------------------------------
// one bitonic compare-exchange pass over both sort buffers (gridDim.y selects)
// ---------------------------------------------------------------------------
__global__ void k_bitonic(float* ws, int j, int k) {
    float* data = (blockIdx.y == 0) ? (ws + WS_PSORT) : (ws + WS_GSORT);
    const unsigned idx = blockIdx.x * blockDim.x + threadIdx.x;  // 0..TOT_-1
    const unsigned b   = idx >> 17;                              // batch
    const unsigned i   = idx & (SORTN - 1);
    float* d           = data + ((size_t)b << 17);
    const unsigned ixj = i ^ (unsigned)j;
    if (ixj > i) {
        float a = d[i];
        float c = d[ixj];
        bool up = ((i & (unsigned)k) == 0);
        if (up ? (a > c) : (a < c)) {
            d[i]   = c;
            d[ixj] = a;
        }
    }
}

// ---------------------------------------------------------------------------
// per-batch EMD partial sums: mean |sort(p) - sort(g)|
// ---------------------------------------------------------------------------
__global__ void k_emd(float* ws) {
    const float* ps = ws + WS_PSORT;
    const float* gs = ws + WS_GSORT;
    const unsigned idx = blockIdx.x * blockDim.x + threadIdx.x;  // exactly TOT_
    const unsigned b   = idx >> 17;       // all 32 lanes of a wave share b
    float dfr = fabsf(ps[idx] - gs[idx]);
    dfr = wave_sum(dfr);
    if ((threadIdx.x & 31) == 0) atomicAdd(&ws[WS_SCAL + 3 + b], dfr);
}

// ---------------------------------------------------------------------------
// finalize: out[b] = mae + chamfer + emd[b]   (ALPHA = BETA = GAMMA = 1)
// ---------------------------------------------------------------------------
__global__ void k_finalize(const float* ws, float* out) {
    const int b = threadIdx.x;
    if (b < B_) {
        float mae = ws[WS_SCAL + 0] * (1.0f / (float)TOT_);
        float ch  = ws[WS_SCAL + 1] * (1.0f / (float)BN_) +
                    ws[WS_SCAL + 2] * (1.0f / (float)NM_);
        float emd = ws[WS_SCAL + 3 + b] * (1.0f / (float)SORTN);
        out[b] = mae + ch + emd;
    }
}

// ---------------------------------------------------------------------------
extern "C" void kernel_launch(void* const* d_in, const int* in_sizes, int n_in,
                              void* d_out, int out_size, void* d_ws, size_t ws_size,
                              hipStream_t stream) {
    const float* pred = (const float*)d_in[0];
    const float* targ = (const float*)d_in[1];
    float* out = (float*)d_out;
    float* ws  = (float*)d_ws;

    // 1) init accumulators + inf-fill min buffers
    k_init<<<NM_ / 256, 256, 0, stream>>>(ws);

    // 2) row norms (fp32-exact)
    k_norms<<<(2 * BN_) / 256, 256, 0, stream>>>(pred, targ, ws);

    // 3) streaming prep: MAE + sort copies + bf16 hi/lo split
    k_prep<<<(TOT_ / 4) / 256, 256, 0, stream>>>(pred, targ, ws);

    // 4) chamfer distance tiles (WMMA bf16x3 on pre-split data)
    k_chamfer_wmma<<<dim3(N_ / 16, N_ / 64), 128, 0, stream>>>(ws);

    // 5) reduce the two chamfer min-buffers (colmin stored as monotone uint bits)
    k_sum<<<128, 256, 0, stream>>>(ws + WS_COLMIN, BN_, &ws[WS_SCAL + 1]);
    k_sum<<<1024, 256, 0, stream>>>(ws + WS_BATCHMIN, NM_, &ws[WS_SCAL + 2]);

    // 6) bitonic sort both buffers per batch (2^17 elements -> 153 passes)
    for (int k = 2; k <= SORTN; k <<= 1)
        for (int j = k >> 1; j > 0; j >>= 1)
            k_bitonic<<<dim3(TOT_ / 256, 2), 256, 0, stream>>>(ws, j, k);

    // 7) per-batch EMD sums
    k_emd<<<TOT_ / 256, 256, 0, stream>>>(ws);

    // 8) combine
    k_finalize<<<1, 32, 0, stream>>>(ws, out);
}
